// ASTT_GCN_18090402251003
// MI455X (gfx1250) — matrix operand order repaired and checked by reading, not compiled, not run on hardware
//
#include <hip/hip_runtime.h>
#include <hip/hip_bf16.h>

typedef _Float16 h16;
typedef __attribute__((ext_vector_type(16))) _Float16 v16h;
typedef __attribute__((ext_vector_type(8)))  float    v8f;

#define B_   4
#define T_   512
#define K_   68
#define C_   256
#define H_   4
#define HC_  64
#define ROWS_  (B_*T_*K_)         // 139264
#define NELEM_ ((size_t)ROWS_*C_) // 35651584
#define EPS_ 1e-5f

// ---------------------------------------------------------------------------
// Tensor Data Mover: DMA a contiguous 16 KB (4096 f32) tile global -> LDS.
// D# layout per CDNA5 ISA ch.8: group0 = {count, lds_addr, global_addr, type},
// group1 = {data_size, tensor dims/strides, tile dims}. 1-D tile:
// tile_dim0 = 4096 elems (4B), tensor_dim0 = stride0 = 4096, tile_dim1 = 1.
// This toolchain exposes the 6-arg builtin:
//   (u32x4 g0, i32x8 g1, i32x4 g2, i32x4 g3, i32x8 g4, i32 cpol)
// ---------------------------------------------------------------------------
#if __has_builtin(__builtin_amdgcn_tensor_load_to_lds) && __has_builtin(__builtin_amdgcn_s_wait_tensorcnt)
#define HAVE_TDM 1
typedef __attribute__((ext_vector_type(4))) unsigned int u32x4;
typedef __attribute__((ext_vector_type(8))) int          i32x8;
typedef __attribute__((ext_vector_type(4))) int          i32x4;

__device__ __forceinline__ void tdm_load_16kb(const float* gsrc, unsigned lds_addr) {
    unsigned long long ga = (unsigned long long)(uintptr_t)gsrc;
    u32x4 g0;
    g0[0] = 1u;                                                // count=1, is_restore=0, gather off
    g0[1] = lds_addr;                                          // lds_addr (bytes)
    g0[2] = (unsigned)ga;                                      // global_addr[31:0]
    g0[3] = (unsigned)((ga >> 32) & 0x1FFFFFFu) | (2u << 30);  // global_addr[56:32] | type=2
    i32x8 g1;
    g1[0] = (int)(2u << 16);          // workgroup_mask=0 (not in cluster), data_size=2 (4B)
    g1[1] = (int)(4096u << 16);       // tensor_dim0[15:0] << 16 (bits 63:48)
    g1[2] = (int)(1u << 16);          // tensor_dim0[31:16]=0 | tensor_dim1[15:0]=1 << 16
    g1[3] = (int)(4096u << 16);       // tensor_dim1[31:16]=0 | tile_dim0=4096 << 16
    g1[4] = 1;                        // tile_dim1=1, tile_dim2=0
    g1[5] = 4096;                     // tensor_dim0_stride[31:0]
    g1[6] = 0;                        // stride0 hi | stride1 lo
    g1[7] = 0;                        // stride1 hi
    i32x4 z4 = {0, 0, 0, 0};          // groups 2/3 unused (<=2D tensor)
    i32x8 z8 = {0, 0, 0, 0, 0, 0, 0, 0};
    __builtin_amdgcn_tensor_load_to_lds(g0, g1, z4, z4, z8, 0);
}
#endif

// ---------------------------------------------------------------------------
// WMMA fragment loaders (wave32, 16x16x32 f16 -> f32)
// A 16x32 f16 layout: lane<16: row=m0+lane, K=k0+0..15
//                     lane>=16: same row, K=k0+16..31 (matches B order)
// B 32x16 f16 layout: lane<16: col=n0+lane, K=k0+0..15; lane>=16: col, K=k0+16..31
// C/D f32:            lane<16: col=n0+lane, VGPRv: row=m0+v; lane>=16: row=m0+8+v
// ---------------------------------------------------------------------------

__device__ __forceinline__ v16h ld_a_h(const h16* base, int ld, int m0, int k0, int lane) {
    int hl = lane >> 4, lr = lane & 15;
    const h16* row = base + (m0 + lr) * ld;
    int ka = k0 + hl * 16, kb = ka + 8;
    v16h a;
#pragma unroll
    for (int e = 0; e < 8; ++e) { a[e] = row[ka + e]; a[8 + e] = row[kb + e]; }
    return a;
}

__device__ __forceinline__ v16h ld_b_h(const h16* base, int ld, int k0, int n0, int lane) {
    int hl = lane >> 4, lr = lane & 15;
    const h16* p = base + (size_t)(k0 + hl * 16) * ld + (n0 + lr);
    v16h b;
#pragma unroll
    for (int e = 0; e < 16; ++e) { b[e] = p[(size_t)e * ld]; }
    return b;
}

// B from f32 global weights (convert inline)
__device__ __forceinline__ v16h ld_b_f32(const float* W, int ld, int k0, int n0, int lane) {
    int hl = lane >> 4, lr = lane & 15;
    const float* p = W + (size_t)(k0 + hl * 16) * ld + (n0 + lr);
    v16h b;
#pragma unroll
    for (int e = 0; e < 16; ++e) { b[e] = (h16)p[(size_t)e * ld]; }
    return b;
}

// A from f32 global x slice (row-guard: rows >= K_ read as 0)
__device__ __forceinline__ v16h ld_a_x(const float* xb, int m0, int k0, int lane) {
    int hl = lane >> 4, lr = lane & 15;
    int r = m0 + lr;
    bool ok = (r < K_);
    const float* row = xb + (size_t)(ok ? r : 0) * C_;
    int ka = k0 + hl * 16, kb = ka + 8;
    v16h a;
#pragma unroll
    for (int e = 0; e < 8; ++e) {
        a[e]     = ok ? (h16)row[ka + e] : (h16)0.0f;
        a[8 + e] = ok ? (h16)row[kb + e] : (h16)0.0f;
    }
    return a;
}

// B = k^T from LDS k (row-major [kj][c]): B[kdim=c][n=kj] = k[kj][c]
__device__ __forceinline__ v16h ld_bT_h(const h16* base, int ld, int k0, int n0, int lane) {
    int hl = lane >> 4, lr = lane & 15;
    const h16* row = base + (size_t)(n0 + lr) * ld + k0 + hl * 16;
    v16h b;
#pragma unroll
    for (int e = 0; e < 16; ++e) { b[e] = row[e]; }
    return b;
}

// A from attention-probability byte buffer (row stride 320B, 96 valid f16/row)
__device__ __forceinline__ v16h ld_a_att(const unsigned char* sa, int m0, int k0, int lane) {
    int hl = lane >> 4, lr = lane & 15;
    const h16* row = (const h16*)(sa + (size_t)(m0 + lr) * 320);
    int ka = k0 + hl * 16, kb = ka + 8;
    v16h a;
#pragma unroll
    for (int e = 0; e < 8; ++e) { a[e] = row[ka + e]; a[8 + e] = row[kb + e]; }
    return a;
}

#define WMMA(a, b, c) __builtin_amdgcn_wmma_f32_16x16x32_f16(false, (a), false, (b), (short)0, (c), false, false)

// ---------------------------------------------------------------------------
// Kernel 1: fused per-(b,t,h) attention.  grid = B*T*H blocks of 32 threads.
// ---------------------------------------------------------------------------
__device__ __forceinline__ void proj_head(const float* xb, const float* W, const float* bias,
                                          h16* out_s, int mt, int lane) {
    v8f acc[4];
#pragma unroll
    for (int nt = 0; nt < 4; ++nt) acc[nt] = {};
#pragma unroll
    for (int kc = 0; kc < 2; ++kc) {
        v16h a = ld_a_x(xb, mt * 16, kc * 32, lane);
#pragma unroll
        for (int nt = 0; nt < 4; ++nt) {
            v16h b = ld_b_f32(W, HC_, kc * 32, nt * 16, lane);
            acc[nt] = WMMA(a, b, acc[nt]);
        }
    }
    int hl = lane >> 4, lr = lane & 15;
#pragma unroll
    for (int nt = 0; nt < 4; ++nt) {
        int col = nt * 16 + lr;
        float bv = bias[col];
#pragma unroll
        for (int v = 0; v < 8; ++v) {
            int row = mt * 16 + hl * 8 + v;  // <= 79
            out_s[row * HC_ + col] = (h16)(acc[nt][v] + bv);
        }
    }
}

__global__ __launch_bounds__(32)
void attn_kernel(const float* __restrict__ x,
                 const float* __restrict__ Wq, const float* __restrict__ bq,
                 const float* __restrict__ Wk, const float* __restrict__ bk,
                 const float* __restrict__ Wv, const float* __restrict__ bv,
                 const float* __restrict__ a_se,
                 float* __restrict__ y1, float* __restrict__ sum1, float* __restrict__ sq1) {
    __shared__ h16 q_s[80 * HC_];
    __shared__ h16 k_s[80 * HC_];
    __shared__ h16 v_s[96 * HC_];
    __shared__ __align__(16) unsigned char sa[80 * 320]; // scores f32 (80 cols) overlaid with att f16 (96 cols)
    __shared__ float bns[HC_], bnss[HC_];

    int lane = threadIdx.x;
    int bt = blockIdx.x >> 2;
    int h  = blockIdx.x & 3;
    const float* xb = x + (size_t)bt * K_ * C_ + h * HC_;

    for (int i = lane; i < 80 * HC_; i += 32) { q_s[i] = (h16)0.0f; k_s[i] = (h16)0.0f; }
    for (int i = lane; i < 96 * HC_; i += 32) { v_s[i] = (h16)0.0f; }
    for (int i = lane; i < HC_; i += 32) { bns[i] = 0.0f; bnss[i] = 0.0f; }
    __syncthreads();

    // ---- Q, K, V projections (M=68->80, N=64, K=64) ----
    for (int mt = 0; mt < 5; ++mt) {
        proj_head(xb, Wq, bq, q_s, mt, lane);
        proj_head(xb, Wk, bk, k_s, mt, lane);
        proj_head(xb, Wv, bv, v_s, mt, lane);
    }
    __syncthreads();

    // ---- scores = q @ k^T  (M=80, N=80, K=64) ----
    int hl = lane >> 4, lr = lane & 15;
    for (int mt = 0; mt < 5; ++mt) {
        v8f acc[5];
#pragma unroll
        for (int nt = 0; nt < 5; ++nt) acc[nt] = {};
#pragma unroll
        for (int kc = 0; kc < 2; ++kc) {
            v16h a = ld_a_h(q_s, HC_, mt * 16, kc * 32, lane);
#pragma unroll
            for (int nt = 0; nt < 5; ++nt) {
                v16h b = ld_bT_h(k_s, HC_, kc * 32, nt * 16, lane);
                acc[nt] = WMMA(a, b, acc[nt]);
            }
        }
#pragma unroll
        for (int nt = 0; nt < 5; ++nt) {
            int col = nt * 16 + lr;
#pragma unroll
            for (int v = 0; v < 8; ++v) {
                int row = mt * 16 + hl * 8 + v;  // <= 79, fits buffer
                ((float*)(sa + (size_t)row * 320))[col] = acc[nt][v];
            }
        }
    }
    __syncthreads();

    // ---- softmax over 68 cols, + a_se, convert to f16 in place (96 cols, zero-padded) ----
    for (int r = lane; r < 80; r += 32) {
        volatile float* srow = (volatile float*)(sa + (size_t)r * 320);
        volatile h16*   arow = (volatile h16*)(sa + (size_t)r * 320);
        if (r < K_) {
            float mx = -1e30f;
            for (int c = 0; c < K_; ++c) mx = fmaxf(mx, srow[c]);
            float s = 0.0f;
            for (int c = 0; c < K_; ++c) { float e = __expf(srow[c] - mx); srow[c] = e; s += e; }
            float inv = 1.0f / s;
            for (int c = 0; c < 96; ++c) {
                float val = (c < K_) ? (srow[c] * inv + a_se[r * K_ + c]) : 0.0f;
                arow[c] = (h16)val; // writes bytes 2c..2c+1, f32 elem c/2 already consumed
            }
        } else {
            for (int c = 0; c < 96; ++c) arow[c] = (h16)0.0f;
        }
    }
    __syncthreads();

    // ---- out = att @ v  (M=80, N=64, K=68->96), + residual, BN1 stats ----
    for (int mt = 0; mt < 5; ++mt) {
        v8f acc[4];
#pragma unroll
        for (int nt = 0; nt < 4; ++nt) acc[nt] = {};
#pragma unroll
        for (int kc = 0; kc < 3; ++kc) {
            v16h a = ld_a_att(sa, mt * 16, kc * 32, lane);
#pragma unroll
            for (int nt = 0; nt < 4; ++nt) {
                v16h b = ld_b_h(v_s, HC_, kc * 32, nt * 16, lane);
                acc[nt] = WMMA(a, b, acc[nt]);
            }
        }
#pragma unroll
        for (int nt = 0; nt < 4; ++nt) {
            int col = nt * 16 + lr;
            float s = 0.0f, s2 = 0.0f;
#pragma unroll
            for (int v = 0; v < 8; ++v) {
                int row = mt * 16 + hl * 8 + v;
                if (row < K_) {
                    float val = acc[nt][v] + xb[(size_t)row * C_ + col];  // residual
                    y1[((size_t)bt * K_ + row) * C_ + h * HC_ + col] = val;
                    s += val; s2 += val * val;
                }
            }
            atomicAdd(&bns[col], s);
            atomicAdd(&bnss[col], s2);
        }
    }
    __syncthreads();
    for (int c = lane; c < HC_; c += 32) {
        atomicAdd(&sum1[h * HC_ + c], bns[c]);
        atomicAdd(&sq1[h * HC_ + c], bnss[c]);
    }
}

// ---------------------------------------------------------------------------
// Kernel: BN stats -> per-channel scale/shift (1 block x 256)
// ---------------------------------------------------------------------------
__global__ void bn_finalize(const float* __restrict__ sum, const float* __restrict__ sq,
                            const float* __restrict__ g, const float* __restrict__ b,
                            float* __restrict__ scale, float* __restrict__ shift) {
    int c = threadIdx.x;
    float inv_n = 1.0f / (float)ROWS_;
    float mean = sum[c] * inv_n;
    float var  = sq[c] * inv_n - mean * mean;
    float sc = g[c] * rsqrtf(var + EPS_);
    scale[c] = sc;
    shift[c] = b[c] - mean * sc;
}

// ---------------------------------------------------------------------------
// Kernel: fused FFN.  16-row tile per block, 4 waves. Hidden stays in LDS (f16).
// y1 tile staged into LDS by the Tensor Data Mover (TENSORcnt), residual
// recomputed from the raw LDS tile (no f32 shadow copy).
// ---------------------------------------------------------------------------
__global__ __launch_bounds__(128)
void ffn_kernel(const float* __restrict__ y1,
                const float* __restrict__ scale1, const float* __restrict__ shift1,
                const h16* __restrict__ W1h, const float* __restrict__ bL1,
                const h16* __restrict__ W2h, const float* __restrict__ bL2,
                float* __restrict__ hout, float* __restrict__ sum2, float* __restrict__ sq2) {
    __shared__ __align__(16) float y1s[16 * C_];  // 16 KB raw y1 tile (TDM destination)
    __shared__ h16   xbh[16 * C_];                //  8 KB BN1-normalized tile, f16 (GEMM1 A)
    __shared__ h16   hid[16 * 1024];              // 32 KB ReLU hidden (GEMM2 A)
    __shared__ float bns[C_], bnss[C_];           //  2 KB

    int tid = threadIdx.x;
    int lane = tid & 31;
    int w = tid >> 5;
    size_t row0 = (size_t)blockIdx.x * 16;

#ifdef HAVE_TDM
    if (w == 0) {
        tdm_load_16kb(y1 + row0 * C_, (unsigned)(uintptr_t)&y1s[0]);
        __builtin_amdgcn_s_wait_tensorcnt(0);
    }
#else
    for (int i = tid; i < 16 * C_; i += 128) y1s[i] = y1[row0 * C_ + i];
#endif
    // speculative prefetch of next block's tile toward L2 while we compute
    if (tid == 0) __builtin_prefetch(y1 + (row0 + 16) * C_, 0, 1);
    for (int c = tid; c < C_; c += 128) { bns[c] = 0.0f; bnss[c] = 0.0f; }
    __syncthreads();

    // BN1 normalize -> f16 A operand
    for (int i = tid; i < 16 * C_; i += 128) {
        int c = i & (C_ - 1);
        xbh[i] = (h16)(y1s[i] * scale1[c] + shift1[c]);
    }
    __syncthreads();

    int hl = lane >> 4, lr = lane & 15;

    // GEMM1: (16 x 256) @ (256 x 1024), wave w owns cols [w*256, w*256+256)
    for (int nt = 0; nt < 16; ++nt) {
        int n0 = w * 256 + nt * 16;
        v8f acc = {};
#pragma unroll
        for (int kc = 0; kc < 8; ++kc) {
            v16h a = ld_a_h(xbh, C_, 0, kc * 32, lane);
            v16h b = ld_b_h(W1h, 1024, kc * 32, n0, lane);
            acc = WMMA(a, b, acc);
        }
        int col = n0 + lr;
        float bias = bL1[col];
#pragma unroll
        for (int v = 0; v < 8; ++v) {
            float t = acc[v] + bias;
            t = t > 0.0f ? t : 0.0f;  // ReLU
            hid[(hl * 8 + v) * 1024 + col] = (h16)t;
        }
    }
    __syncthreads();

    // GEMM2: (16 x 1024) @ (1024 x 256), wave w owns cols [w*64, w*64+64)
    for (int nt = 0; nt < 4; ++nt) {
        int n0 = w * 64 + nt * 16;
        v8f acc = {};
        for (int kc = 0; kc < 32; ++kc) {
            v16h a = ld_a_h(hid, 1024, 0, kc * 32, lane);
            v16h b = ld_b_h(W2h, C_, kc * 32, n0, lane);
            acc = WMMA(a, b, acc);
        }
        int col = n0 + lr;
        float bias = bL2[col];
        float sc1 = scale1[col], sh1 = shift1[col];
        float s = 0.0f, s2 = 0.0f;
#pragma unroll
        for (int v = 0; v < 8; ++v) {
            int r = hl * 8 + v;
            float resid = y1s[r * C_ + col] * sc1 + sh1;       // recompute BN1(y1) residual
            float val = acc[v] + bias + resid;
            hout[(row0 + r) * C_ + col] = val;
            s += val; s2 += val * val;
        }
        atomicAdd(&bns[col], s);
        atomicAdd(&bnss[col], s2);
    }
    __syncthreads();
    for (int c = tid; c < C_; c += 128) {
        atomicAdd(&sum2[c], bns[c]);
        atomicAdd(&sq2[c], bnss[c]);
    }
}

// ---------------------------------------------------------------------------
// Small utility kernels
// ---------------------------------------------------------------------------
__global__ void zero_stats_kernel(float* p) {
    p[(size_t)blockIdx.x * 256 + threadIdx.x] = 0.0f;
}

__global__ void cvt_weights_kernel(const float* __restrict__ W1, const float* __restrict__ W2,
                                   h16* __restrict__ W1h, h16* __restrict__ W2h) {
    size_t i = (size_t)blockIdx.x * 256 + threadIdx.x;
    const size_t n1 = (size_t)C_ * 4 * C_;  // 262144
    if (i < n1) W1h[i] = (h16)W1[i];
    else        W2h[i - n1] = (h16)W2[i - n1];
}

__global__ void apply_bn_kernel(const float* __restrict__ hbuf,
                                const float* __restrict__ scale, const float* __restrict__ shift,
                                float* __restrict__ out) {
    size_t i = (size_t)blockIdx.x * 256 + threadIdx.x;
    int c = threadIdx.x;  // blockDim=256 == C_, row-major -> channel == tid
    out[i] = hbuf[i] * scale[c] + shift[c];
}

// ---------------------------------------------------------------------------
// Host launcher
// ---------------------------------------------------------------------------
extern "C" void kernel_launch(void* const* d_in, const int* in_sizes, int n_in,
                              void* d_out, int out_size, void* d_ws, size_t ws_size,
                              hipStream_t stream) {
    const float* x    = (const float*)d_in[0];
    const float* Wq   = (const float*)d_in[1];
    const float* bq   = (const float*)d_in[2];
    const float* Wk   = (const float*)d_in[3];
    const float* bk   = (const float*)d_in[4];
    const float* Wv   = (const float*)d_in[5];
    const float* bv   = (const float*)d_in[6];
    const float* a_se = (const float*)d_in[7];
    const float* g1   = (const float*)d_in[8];
    const float* b1   = (const float*)d_in[9];
    const float* W1   = (const float*)d_in[10];
    const float* bL1  = (const float*)d_in[11];
    const float* W2   = (const float*)d_in[12];
    const float* bL2  = (const float*)d_in[13];
    const float* g2   = (const float*)d_in[14];
    const float* b2   = (const float*)d_in[15];

    // workspace layout
    float* y1    = (float*)d_ws;          // NELEM_
    float* hbuf  = y1 + NELEM_;           // NELEM_
    float* stats = hbuf + NELEM_;         // 2048 floats
    float* sum1 = stats, *sq1 = stats + 256, *sum2 = stats + 512, *sq2 = stats + 768;
    float* scale1 = stats + 1024, *shift1 = stats + 1280;
    float* scale2 = stats + 1536, *shift2 = stats + 1792;
    h16* W1h = (h16*)(stats + 2048);      // 262144 halves
    h16* W2h = W1h + (size_t)C_ * 4 * C_; // 262144 halves

    zero_stats_kernel<<<4, 256, 0, stream>>>(stats);                       // zero sums (1024 floats)
    cvt_weights_kernel<<<2048, 256, 0, stream>>>(W1, W2, W1h, W2h);

    attn_kernel<<<B_ * T_ * H_, 32, 0, stream>>>(x, Wq, bq, Wk, bk, Wv, bv, a_se, y1, sum1, sq1);
    bn_finalize<<<1, 256, 0, stream>>>(sum1, sq1, g1, b1, scale1, shift1);

    ffn_kernel<<<ROWS_ / 16, 128, 0, stream>>>(y1, scale1, shift1, W1h, bL1, W2h, bL2, hbuf, sum2, sq2);
    bn_finalize<<<1, 256, 0, stream>>>(sum2, sq2, g2, b2, scale2, shift2);

    apply_bn_kernel<<<ROWS_, 256, 0, stream>>>(hbuf, scale2, shift2, (float*)d_out);
}
